// QRNNLayer_22170621182259
// MI455X (gfx1250) — matrix operand order repaired and checked by reading, not compiled, not run on hardware
//
#include <hip/hip_runtime.h>
#include <hip/hip_bf16.h>

// ---------------------------------------------------------------------------
// QRNN layer for MI455X (gfx1250, wave32, WMMA + async-to-LDS).
// conv1d(K=2, causal) as bf16 WMMA GEMM (M=16384, N=3072, K=2048) with fused
// bias + tanh/sigmoid epilogue, followed by the sequential fo-pool scan.
// GEMM tiles are DMA'd into LDS with GLOBAL_LOAD_ASYNC_TO_LDS_B128 (ASYNCcnt),
// double-buffered with one barrier per K-step.
// ---------------------------------------------------------------------------

typedef __attribute__((ext_vector_type(16))) __bf16 v16bf;
typedef __attribute__((ext_vector_type(8)))  float  v8f;

union FragAB { uint4 u[2]; v16bf v; };

// fp32 -> bf16 bits, round-to-nearest-even
static __device__ __forceinline__ unsigned short f2bf(float x) {
    union { float f; unsigned u; } v; v.f = x;
    unsigned r = v.u + 0x7FFFu + ((v.u >> 16) & 1u);
    return (unsigned short)(r >> 16);
}

// generic shared-pointer -> raw LDS byte offset (low 32 bits of aperture addr)
static __device__ __forceinline__ unsigned lds_off(const void* p) {
    return (unsigned)(unsigned long long)p;
}

static __device__ __forceinline__ void async_b128(unsigned ldsaddr,
                                                  unsigned voff,
                                                  const void* sbase) {
    asm volatile("global_load_async_to_lds_b128 %0, %1, %2"
                 :: "v"(ldsaddr), "v"(voff), "s"(sbase)
                 : "memory");
}

static __device__ __forceinline__ void wait_async0() {
    asm volatile("s_wait_asynccnt 0" ::: "memory");
}

// ---------------- pack x: (B,S,D) fp32 -> bf16, 4 elems/thread -------------
__global__ void pack_x_kernel(const float* __restrict__ x,
                              unsigned short* __restrict__ xb) {
    int i = blockIdx.x * 256 + threadIdx.x;          // over 16777216/4 groups
    size_t base = (size_t)i * 4;
    float4 v = *(const float4*)(x + base);
    ushort4 o;
    o.x = f2bf(v.x); o.y = f2bf(v.y); o.z = f2bf(v.z); o.w = f2bf(v.w);
    *(ushort4*)(xb + base) = o;
}

// ------- pack w: (3H, D, 2) fp32 -> Wt (3H, 2D) bf16, k-major rows ---------
// Wt[n][k] = (k < 1024) ? w[n][k][0] : w[n][k-1024][1]
__global__ void pack_w_kernel(const float* __restrict__ w,
                              unsigned short* __restrict__ wt) {
    int i = blockIdx.x * 256 + threadIdx.x;          // over 3072*2048
    int n = i >> 11;
    int k = i & 2047;
    size_t src = (size_t)n * 2048 + ((k < 1024) ? (size_t)k * 2
                                                : (size_t)(k - 1024) * 2 + 1);
    wt[i] = f2bf(w[src]);
}

// ---------------------------- GEMM + activations ---------------------------
// Tile: 128(M) x 128(N), 8 waves: waveM in 0..3 (x32 rows), waveN in 0..1
// (x64 cols). Each wave: 2x4 WMMA 16x16x32 bf16 tiles. K-step 64 via
// async-to-LDS double buffering (one barrier per step).
#define LDP 72   // padded LDS row stride (elements) -> 144B, 16B aligned

__global__ __launch_bounds__(256)
void qrnn_gemm_kernel(const unsigned short* __restrict__ Xb,  // (B,S,D) bf16
                      const unsigned short* __restrict__ Wt,  // (3072,2048) bf16
                      const float* __restrict__ bias,         // (3072,)
                      float* __restrict__ gates)              // (16384,3072) f32
{
    __shared__ __align__(16) unsigned short Asm[2][128 * LDP];
    __shared__ __align__(16) unsigned short Bsm[2][128 * LDP];

    const int t     = threadIdx.x;
    const int lane  = t & 31;
    const int wv    = t >> 5;
    const int waveM = wv & 3;
    const int waveN = wv >> 2;

    const int m0 = blockIdx.x * 128;      // 128 M-tiles
    const int n0 = blockIdx.y * 128;      // 24  N-tiles
    const int b  = m0 >> 11;              // batch (2048 rows per batch)
    const int s0 = m0 & 2047;

    const int lrow  = t >> 1;             // 0..127 : tile row loaded
    const int lhalf = t & 1;              // 0/1    : which 32-elem k half

    // per-thread LDS chunk offsets (bytes) for each buffer
    const unsigned ldsA0 = lds_off(&Asm[0][lrow * LDP + lhalf * 32]);
    const unsigned ldsA1 = lds_off(&Asm[1][lrow * LDP + lhalf * 32]);
    const unsigned ldsB0 = lds_off(&Bsm[0][lrow * LDP + lhalf * 32]);
    const unsigned ldsB1 = lds_off(&Bsm[1][lrow * LDP + lhalf * 32]);
    const unsigned wrowB = (unsigned)((n0 + lrow) * 2048 + lhalf * 32) * 2u;

    // issue one K-tile (kc) into buffer buf via async DMA to LDS
    auto issue_tile = [&](int kc, int buf) {
        const int tap1 = (kc >= 1024);
        const int srow = s0 + lrow - (tap1 ? 0 : 1);     // x[s-1] / x[s]
        const int koff = (tap1 ? (kc - 1024) : kc) + lhalf * 32;
        const unsigned la = buf ? ldsA1 : ldsA0;
        const unsigned lb = buf ? ldsB1 : ldsB0;
        if (srow >= 0) {
            const unsigned voff =
                (unsigned)(((b * 2048 + srow) * 1024 + koff) * 2);
#pragma unroll
            for (int j = 0; j < 4; ++j)
                async_b128(la + j * 16u, voff + j * 16u, (const void*)Xb);
        } else {                       // causal zero row (s==0, tap 0)
            uint4 z; z.x = z.y = z.z = z.w = 0u;
            uint4* da = (uint4*)&Asm[buf][lrow * LDP + lhalf * 32];
#pragma unroll
            for (int j = 0; j < 4; ++j) da[j] = z;
        }
        const unsigned voffB = wrowB + (unsigned)(kc * 2);
#pragma unroll
        for (int j = 0; j < 4; ++j)
            async_b128(lb + j * 16u, voffB + j * 16u, (const void*)Wt);
    };

    v8f acc[2][4] = {};

    issue_tile(0, 0);                     // prologue DMA of tile 0

    for (int it = 0; it < 32; ++it) {     // 2048 / 64 K-steps
        const int cur = it & 1;
        wait_async0();                    // my chunks of tile `it` landed
        __syncthreads();                  // tile `it` visible; prev reads done
        if (it + 1 < 32)
            issue_tile((it + 1) * 64, cur ^ 1);   // DMA next tile (overlaps)

        const unsigned short* As = &Asm[cur][0];
        const unsigned short* Bs = &Bsm[cur][0];

#pragma unroll
        for (int ks = 0; ks < 64; ks += 32) {
            FragAB a[2], bf[4];
            // A 16x32 bf16 layout: lanes 0-15 rows, K {kp..kp+7, 16+kp..}
#pragma unroll
            for (int mt = 0; mt < 2; ++mt) {
                const int row = waveM * 32 + mt * 16 + (lane & 15);
                const int kp  = (lane >> 4) * 8;
                const unsigned short* p = As + row * LDP + ks + kp;
                a[mt].u[0] = *(const uint4*)p;
                a[mt].u[1] = *(const uint4*)(p + 16);
            }
            // B 32x16 bf16 layout: lane%16 = col, K {kb..kb+7, 16+kb..}
#pragma unroll
            for (int nt = 0; nt < 4; ++nt) {
                const int col = waveN * 64 + nt * 16 + (lane & 15);
                const int kb  = ks + (lane >> 4) * 8;
                const unsigned short* p = Bs + col * LDP + kb;
                bf[nt].u[0] = *(const uint4*)p;
                bf[nt].u[1] = *(const uint4*)(p + 16);
            }
#pragma unroll
            for (int mt = 0; mt < 2; ++mt)
#pragma unroll
                for (int nt = 0; nt < 4; ++nt)
                    acc[mt][nt] = __builtin_amdgcn_wmma_f32_16x16x32_bf16(
                        false, a[mt].v, false, bf[nt].v,
                        (short)0, acc[mt][nt], false, false);
        }
    }

    // ---- epilogue: bias + gate activation, fp32 store ----
    // C layout: VGPR r, lane: n = lane%16, m = r + (lane>=16 ? 8 : 0)
#pragma unroll
    for (int mt = 0; mt < 2; ++mt) {
#pragma unroll
        for (int nt = 0; nt < 4; ++nt) {
            const int ng = n0 + waveN * 64 + nt * 16 + (lane & 15);
            const float bv = bias[ng];
            const int chunk = ng >> 10;             // 0:Z 1:F 2:O
#pragma unroll
            for (int r = 0; r < 8; ++r) {
                const int mg = m0 + waveM * 32 + mt * 16 + r + ((lane >> 4) * 8);
                float v = acc[mt][nt][r] + bv;
                float g = (chunk == 0) ? tanhf(v)
                                       : 1.0f / (1.0f + __expf(-v));
                gates[(size_t)mg * 3072 + ng] = g;
            }
        }
    }
}

// ------------------------------ fo-pool scan -------------------------------
__global__ void qrnn_scan_kernel(const float* __restrict__ gates,
                                 float* __restrict__ hout,
                                 float* __restrict__ clast) {
    const int idx = blockIdx.x * blockDim.x + threadIdx.x;   // 8192 = B*H
    const int b = idx >> 10;
    const int h = idx & 1023;
    const float* g = gates + (size_t)b * 2048 * 3072 + h;
    float* o = hout + (size_t)b * 2048 * 1024 + h;
    float c = 0.0f;
#pragma unroll 4
    for (int s = 0; s < 2048; ++s) {
        const float* gs = g + (size_t)s * 3072;
        const float z  = gs[0];
        const float f  = gs[1024];
        const float oo = gs[2048];
        c = f * c + (1.0f - f) * z;
        o[(size_t)s * 1024] = oo * c;
    }
    clast[idx] = c;
}

// ------------------------------- launcher ----------------------------------
extern "C" void kernel_launch(void* const* d_in, const int* in_sizes, int n_in,
                              void* d_out, int out_size, void* d_ws, size_t ws_size,
                              hipStream_t stream) {
    const float* x    = (const float*)d_in[0];   // (8,2048,1024)
    const float* w    = (const float*)d_in[1];   // (3072,1024,2)
    const float* bias = (const float*)d_in[2];   // (3072,)
    float* out = (float*)d_out;                  // h (8,2048,1024) ++ c (8,1024)

    // workspace: Xbf (32MB) | Wt bf16 (12MB) | gates f32 (192MB)
    unsigned short* Xb = (unsigned short*)d_ws;
    unsigned short* Wt = (unsigned short*)((char*)d_ws + (size_t)33554432);
    float* gates = (float*)((char*)d_ws + (size_t)33554432 + 12582912);

    pack_x_kernel<<<16384, 256, 0, stream>>>(x, Xb);     // 16777216/4 elems
    pack_w_kernel<<<24576, 256, 0, stream>>>(w, Wt);     // 6291456 elems

    dim3 grid(128, 24);                                  // (M/128, N/128)
    qrnn_gemm_kernel<<<grid, 256, 0, stream>>>(Xb, Wt, bias, gates);

    qrnn_scan_kernel<<<32, 256, 0, stream>>>(gates, out, out + 16777216);
}
